// CLAHE_87625922773270
// MI455X (gfx1250) — matrix operand order repaired and checked by reading, not compiled, not run on hardware
//
#include <hip/hip_runtime.h>

typedef __attribute__((ext_vector_type(16))) _Float16 v16h;
typedef __attribute__((ext_vector_type(8)))  float    v8f;

#define HH     4096
#define WW     4096
#define NB     256
#define WAVES  8   // one wave32 per 16x16 tile, 8 tiles per block

__global__ __launch_bounds__(256) void clahe_wmma_kernel(
    const float* __restrict__ x,
    const float* __restrict__ mk,
    float* __restrict__ out)
{
    __shared__ __align__(16) unsigned int histU[WAVES * NB];   // per-wave private histograms
    __shared__ _Float16 rowsumH[WAVES * 16];
    __shared__ float    mappedF[WAVES * NB];                   // per-tile LUT for final gather
    __shared__ float    sigF[NB];                              // sigmoid(mapping_kernel), block-shared

    const int tid  = threadIdx.x;
    const int warp = tid >> 5;
    const int lane = tid & 31;

    // Block-shared sigmoid LUT (one exp per thread instead of 8 per lane)
    sigF[tid] = 1.0f / (1.0f + __expf(-mk[tid]));

    // Zero this wave's private histogram (8 bins/lane, 2x uint4)
    {
        uint4 z = {0u, 0u, 0u, 0u};
        uint4* hz = (uint4*)&histU[warp * NB + lane * 8];
        hz[0] = z; hz[1] = z;
    }
    __syncthreads();   // only needed for sigF visibility later

    const int tile = blockIdx.x * WAVES + warp;     // 65536 tiles
    const int tr = tile >> 8;
    const int tc = tile & 255;
    const long long tb = (long long)(tr * 16) * WW + tc * 16;

    // ---- Load 8 pixels per lane (2 x float4, coalesced 64B rows) ----
    float4 px[2];
#pragma unroll
    for (int i = 0; i < 2; ++i) {
        int f   = i * 128 + lane * 4;   // flat pixel index in tile
        int row = f >> 4;
        int col = f & 15;
        px[i] = *(const float4*)(x + tb + (long long)row * WW + col);
    }

    // ---- Histogram into per-wave LDS bins ----
    unsigned int* hb = &histU[warp * NB];
#pragma unroll
    for (int i = 0; i < 2; ++i) {
        float vals[4] = {px[i].x, px[i].y, px[i].z, px[i].w};
#pragma unroll
        for (int j = 0; j < 4; ++j) {
            int b = (int)(vals[j] * (256.0f / 255.0f));   // floor for v >= 0
            b = b < 0 ? 0 : (b > 255 ? 255 : b);
            atomicAdd(&hb[b], 1u);                        // ds_add_u32 (no return)
        }
    }
    asm volatile("s_wait_dscnt 0" ::: "memory");

    const int  r  = lane & 15;
    const bool hi = lane >= 16;

    // ---- A1: clipped histogram as 16x16 f16 A-matrix (K padded to 32) ----
    // A layout: lanes 0-15 hold K=0..7 (V0..3) / K=16..23 (V4..7); lanes 16-31: K=8..15 / K=24..31
    v16h a1;
#pragma unroll
    for (int v = 0; v < 4; ++v) {
        int k0 = 2 * v + (hi ? 8 : 0);
        float h0 = fminf((float)hb[r * 16 + k0],     4.0f);   // CLIP_LIMIT*mean == 4.0 exactly
        float h1 = fminf((float)hb[r * 16 + k0 + 1], 4.0f);
        a1[2 * v]     = (_Float16)h0;
        a1[2 * v + 1] = (_Float16)h1;
    }
#pragma unroll
    for (int v = 4; v < 8; ++v) { a1[2 * v] = (_Float16)0.f; a1[2 * v + 1] = (_Float16)0.f; }

    // ---- B1: upper-triangular ones U[j][c] = (j <= c), j < 16 ----
    // B layout: lanes 0-15 hold K=0..7 (V0..3) / K=16..23 (V4..7, zero pad); lanes 16-31: K=8..15 / K=24..31 (zero pad)
    v16h b1;
#pragma unroll
    for (int v = 0; v < 8; ++v) {
        int j0 = 2 * v + (hi ? 8 : 0);
        b1[2 * v]     = (_Float16)(((v < 4) && (j0     <= r)) ? 1.0f : 0.0f);
        b1[2 * v + 1] = (_Float16)(((v < 4) && (j0 + 1 <= r)) ? 1.0f : 0.0f);
    }

    v8f c = {};
    // c[r][c] = inclusive row scan of clipped histogram
    c = __builtin_amdgcn_wmma_f32_16x16x32_f16(false, a1, false, b1, (short)0, c, false, false);

    // ---- Row sums (clipped), shared via LDS for B2 ----
    float rs = 0.0f;
#pragma unroll
    for (int k = 0; k < 16; ++k)
        rs += fminf((float)hb[r * 16 + k], 4.0f);
    rowsumH[warp * 16 + r] = (_Float16)rs;    // lanes l and l+16 write identical values
    asm volatile("s_wait_dscnt 0" ::: "memory");

    // ---- A2: strict lower-triangular ones L[r][i] = (i < r) ----
    v16h a2;
#pragma unroll
    for (int v = 0; v < 4; ++v) {
        int k0 = 2 * v + (hi ? 8 : 0);
        a2[2 * v]     = (_Float16)((k0     < r) ? 1.0f : 0.0f);
        a2[2 * v + 1] = (_Float16)((k0 + 1 < r) ? 1.0f : 0.0f);
    }
#pragma unroll
    for (int v = 4; v < 8; ++v) { a2[2 * v] = (_Float16)0.f; a2[2 * v + 1] = (_Float16)0.f; }

    // ---- B2: rowsum[i] broadcast across columns ----
    v16h b2;
#pragma unroll
    for (int v = 0; v < 8; ++v) {
        _Float16 s0 = rowsumH[warp * 16 + 2 * (v & 3) + (hi ? 8 : 0)];
        _Float16 s1 = rowsumH[warp * 16 + 2 * (v & 3) + (hi ? 8 : 0) + 1];
        b2[2 * v]     = (v >= 4) ? (_Float16)0.f : s0;
        b2[2 * v + 1] = (v >= 4) ? (_Float16)0.f : s1;
    }

    // c[r][c] += sum_{i<r} rowsum[i]  ->  full 256-bin inclusive prefix of clipped hist
    c = __builtin_amdgcn_wmma_f32_16x16x32_f16(false, a2, false, b2, (short)0, c, false, false);

    // ---- Normalize: redist adds uniform e per bin; cdf[n] = c[n] + (n+1)*e ----
    // C layout: VGPR v, lane l: M = v + 8*(l>=16), N = l&15, n = M*16+N
    float totalClipped = __shfl(c[7], 31);                       // element n=255
    float e    = (256.0f - totalClipped) * (1.0f / 256.0f);      // excess / N_BINS
    float cdf0 = __shfl(c[0], 0) + e;                            // element n=0 (== cdf_min)
    float cdfLast = totalClipped + 256.0f * e;                   // == 256 (cdf_max)
    float denom   = fmaxf(cdfLast - cdf0, 1e-7f);
    float scale   = 255.0f / denom;

    float* mb = &mappedF[warp * NB];
#pragma unroll
    for (int v = 0; v < 8; ++v) {
        int M = v + (hi ? 8 : 0);
        int n = M * 16 + r;
        float cdf = c[v] + (float)(n + 1) * e;
        mb[n] = (cdf - cdf0) * scale * sigF[n];
    }
    asm volatile("s_wait_dscnt 0" ::: "memory");

    // ---- Remap pixels via per-tile LUT, coalesced float4 stores ----
#pragma unroll
    for (int i = 0; i < 2; ++i) {
        int f   = i * 128 + lane * 4;
        int row = f >> 4;
        int col = f & 15;
        float vals[4] = {px[i].x, px[i].y, px[i].z, px[i].w};
        float res[4];
#pragma unroll
        for (int j = 0; j < 4; ++j) {
            int g = (int)vals[j];                  // truncation, matches tf.cast
            g = g < 0 ? 0 : (g > 255 ? 255 : g);
            res[j] = mb[g];
        }
        float4 o; o.x = res[0]; o.y = res[1]; o.z = res[2]; o.w = res[3];
        *(float4*)(out + tb + (long long)row * WW + col) = o;
    }
}

extern "C" void kernel_launch(void* const* d_in, const int* in_sizes, int n_in,
                              void* d_out, int out_size, void* d_ws, size_t ws_size,
                              hipStream_t stream) {
    const float* x  = (const float*)d_in[0];
    const float* mk = (const float*)d_in[1];
    float* out = (float*)d_out;

    const int n_tiles = (HH / 16) * (WW / 16);     // 65536
    const int blocks  = n_tiles / WAVES;           // 8192 blocks x 256 threads
    clahe_wmma_kernel<<<blocks, 256, 0, stream>>>(x, mk, out);
}
